// metapath_classifier_8048768713044
// MI455X (gfx1250) — matrix operand order repaired and checked by reading, not compiled, not run on hardware
//
#include <hip/hip_runtime.h>
#include <hip/hip_bf16.h>
#include <cstdint>

// ---------------------------------------------------------------------------
// Problem constants (match reference)
// ---------------------------------------------------------------------------
#define NA   4096
#define NP   100000
#define EAP  500000
#define EPA  500000
#define DD   128
#define DFF  2048
#define OUTD 16
#define SM   (NA * 2)          // fused sequence rows = NA tokens x 2 metapaths

typedef __attribute__((ext_vector_type(16))) _Float16 v16h;
typedef __attribute__((ext_vector_type(8)))  float    v8f;

__device__ __forceinline__ v8f wmma_f16(v16h a, v16h b, v8f c) {
    // probe-verified: v_wmma_f32_16x16x32_f16, 8-arg form
    return __builtin_amdgcn_wmma_f32_16x16x32_f16(false, a, false, b, (short)0, c, false, false);
}

union FragH { v16h v; _Float16 h[16]; float4 q[2]; };
union FragF { v8f  v; float    f[8];  };
union Pack8 { _Float16 h[8]; float4 q; };

__device__ __forceinline__ void atomic_add_f(float* p, float v) {
    unsafeAtomicAdd(p, v);   // native global_atomic_add_f32 on gfx1250
}

// ---------------------------------------------------------------------------
// Direct-from-global fragment builders (f32 source -> f16 fragment).
// A-fragment (16xK tile, row-major source, row stride `stride` floats):
//   lane holds row M=lane%16; half j holds K=(j&7)+(j>>3)*16+(lane>>4)*8
//   -> per lane: two contiguous 8-float runs at +((lane>>4)*8) and +16+...
// B-fragment (16 cols x 32 K, source row-major [N,K]):
//   lane holds col N=lane%16; half j holds K=(j&7)+(j>>3)*16+(lane>>4)*8
//   -> per lane: two contiguous 8-float runs at +((lane>>4)*8) and +16+...
// ---------------------------------------------------------------------------
__device__ __forceinline__ v16h load_a_frag(const float* __restrict__ base,
                                            size_t stride, int lane, float scale) {
    const int rl = lane & 15;
    const float* p = base + (size_t)rl * stride + ((lane >> 4) << 3);
    float4 u0 = *(const float4*)(p);
    float4 u1 = *(const float4*)(p + 4);
    float4 u2 = *(const float4*)(p + 16);
    float4 u3 = *(const float4*)(p + 20);
    FragH a;
    a.h[0]  = (_Float16)(u0.x * scale); a.h[1]  = (_Float16)(u0.y * scale);
    a.h[2]  = (_Float16)(u0.z * scale); a.h[3]  = (_Float16)(u0.w * scale);
    a.h[4]  = (_Float16)(u1.x * scale); a.h[5]  = (_Float16)(u1.y * scale);
    a.h[6]  = (_Float16)(u1.z * scale); a.h[7]  = (_Float16)(u1.w * scale);
    a.h[8]  = (_Float16)(u2.x * scale); a.h[9]  = (_Float16)(u2.y * scale);
    a.h[10] = (_Float16)(u2.z * scale); a.h[11] = (_Float16)(u2.w * scale);
    a.h[12] = (_Float16)(u3.x * scale); a.h[13] = (_Float16)(u3.y * scale);
    a.h[14] = (_Float16)(u3.z * scale); a.h[15] = (_Float16)(u3.w * scale);
    return a.v;
}

__device__ __forceinline__ v16h load_b_frag(const float* __restrict__ base,
                                            size_t stride, int lane) {
    const int rl = lane & 15;
    const float* p = base + (size_t)rl * stride + ((lane >> 4) << 3);
    float4 u0 = *(const float4*)(p);
    float4 u1 = *(const float4*)(p + 4);
    float4 u2 = *(const float4*)(p + 16);
    float4 u3 = *(const float4*)(p + 20);
    FragH b;
    b.h[0]  = (_Float16)u0.x; b.h[1]  = (_Float16)u0.y;
    b.h[2]  = (_Float16)u0.z; b.h[3]  = (_Float16)u0.w;
    b.h[4]  = (_Float16)u1.x; b.h[5]  = (_Float16)u1.y;
    b.h[6]  = (_Float16)u1.z; b.h[7]  = (_Float16)u1.w;
    b.h[8]  = (_Float16)u2.x; b.h[9]  = (_Float16)u2.y;
    b.h[10] = (_Float16)u2.z; b.h[11] = (_Float16)u2.w;
    b.h[12] = (_Float16)u3.x; b.h[13] = (_Float16)u3.y;
    b.h[14] = (_Float16)u3.z; b.h[15] = (_Float16)u3.w;
    return b.v;
}

// ---------------------------------------------------------------------------
// WMMA GEMM:  C[m][n] = sum_k A[m][k] * W[n][k] + bias[n]   (x @ W.T)
// One wave per 16M x 64N macro-tile (A fragment reused over 4 N-tiles).
// Requires M%16==0, K%32==0, N%16==0 (true for every call in this model).
// ---------------------------------------------------------------------------
__global__ __launch_bounds__(32)
void gemm16_kernel(const float* __restrict__ A, int lda,
                   const float* __restrict__ W,
                   const float* __restrict__ bias,
                   float* __restrict__ C, int ldc,
                   int M, int N, int K, int relu)
{
    const int lane = threadIdx.x;
    const int rl   = lane & 15;
    const int mt   = blockIdx.x * 16;
    const int nt0  = blockIdx.y * 64;

    FragF acc[4];
    #pragma unroll
    for (int t = 0; t < 4; ++t)
        #pragma unroll
        for (int i = 0; i < 8; ++i) acc[t].f[i] = 0.0f;

    for (int k0 = 0; k0 < K; k0 += 32) {
        v16h a = load_a_frag(A + (size_t)mt * lda + k0, lda, lane, 1.0f);
        #pragma unroll
        for (int t = 0; t < 4; ++t) {
            int nr = nt0 + t * 16;
            if (nr < N) {   // uniform (scalar) branch: EXEC untouched
                v16h b = load_b_frag(W + (size_t)nr * K + k0, K, lane);
                acc[t].v = wmma_f16(a, b, acc[t].v);
            }
        }
    }

    const int rowb = mt + ((lane >> 4) << 3);
    #pragma unroll
    for (int t = 0; t < 4; ++t) {
        int col = nt0 + t * 16 + rl;
        if (nt0 + t * 16 < N) {
            float bv = bias ? bias[col] : 0.0f;
            #pragma unroll
            for (int v = 0; v < 8; ++v) {
                float val = acc[t].f[v] + bv;
                if (relu) val = fmaxf(val, 0.0f);
                C[(size_t)(rowb + v) * ldc + col] = val;
            }
        }
    }
}

// ---------------------------------------------------------------------------
// Fold metapath projections: Wf = Wo @ Wv  [128x128],  bf = Wo @ bv + bo
// ---------------------------------------------------------------------------
__global__ void fuse_w_kernel(const float* __restrict__ Wv, const float* __restrict__ bv,
                              const float* __restrict__ Wo, const float* __restrict__ bo,
                              float* __restrict__ Wf, float* __restrict__ bf)
{
    int r = blockIdx.x, c = threadIdx.x;
    float acc = 0.0f;
    for (int k = 0; k < DD; ++k) acc += Wo[r * DD + k] * Wv[k * DD + c];
    Wf[r * DD + c] = acc;
    if (c == 0) {
        float b = bo[r];
        for (int k = 0; k < DD; ++k) b += Wo[r * DD + k] * bv[k];
        bf[r] = b;
    }
}

// ---------------------------------------------------------------------------
// Scatter-mean pieces
// ---------------------------------------------------------------------------
__global__ void count_kernel(const int* __restrict__ dst, float* __restrict__ cnt, int E)
{
    int e = blockIdx.x * blockDim.x + threadIdx.x;
    if (e < E) atomic_add_f(&cnt[dst[e]], 1.0f);
}

__global__ void scatter_add_kernel(const float* __restrict__ proj,
                                   const int* __restrict__ src,
                                   const int* __restrict__ dst,
                                   float* __restrict__ agg, int E)
{
    int t = blockIdx.x * blockDim.x + threadIdx.x;   // 32 threads per edge
    int e = t >> 5;
    if (e >= E) return;
    int q = (t & 31) * 4;
    int s = src[e], d = dst[e];
    float4 v = *(const float4*)(proj + (size_t)s * DD + q);
    float* p = agg + (size_t)d * DD + q;
    atomic_add_f(p + 0, v.x);
    atomic_add_f(p + 1, v.y);
    atomic_add_f(p + 2, v.z);
    atomic_add_f(p + 3, v.w);
}

// ---------------------------------------------------------------------------
// LayerNorm over D=128:  out = LN(X/cnt? + R) * g + b.  One wave per row.
// ---------------------------------------------------------------------------
__global__ void ln_kernel(const float* __restrict__ X, int ldx,
                          const float* __restrict__ R, int ldr,
                          const float* __restrict__ cnt,
                          const float* __restrict__ g, const float* __restrict__ bta,
                          float* __restrict__ out, int ldo, int rows)
{
    int row  = blockIdx.x * (blockDim.x >> 5) + (threadIdx.x >> 5);
    int lane = threadIdx.x & 31;
    if (row >= rows) return;
    float4 xv = *(const float4*)(X + (size_t)row * ldx + lane * 4);
    if (cnt) {
        float inv = 1.0f / fmaxf(cnt[row], 1.0f);
        xv.x *= inv; xv.y *= inv; xv.z *= inv; xv.w *= inv;
    }
    if (R) {
        float4 rv = *(const float4*)(R + (size_t)row * ldr + lane * 4);
        xv.x += rv.x; xv.y += rv.y; xv.z += rv.z; xv.w += rv.w;
    }
    float s = xv.x + xv.y + xv.z + xv.w;
    #pragma unroll
    for (int off = 16; off >= 1; off >>= 1) s += __shfl_xor(s, off, 32);
    float mu = s * (1.0f / 128.0f);
    float4 d = make_float4(xv.x - mu, xv.y - mu, xv.z - mu, xv.w - mu);
    float var = d.x * d.x + d.y * d.y + d.z * d.z + d.w * d.w;
    #pragma unroll
    for (int off = 16; off >= 1; off >>= 1) var += __shfl_xor(var, off, 32);
    float rs = rsqrtf(var * (1.0f / 128.0f) + 1e-5f);
    float4 gv = *(const float4*)(g + lane * 4);
    float4 bv = *(const float4*)(bta + lane * 4);
    float4 ov = make_float4(d.x * rs * gv.x + bv.x, d.y * rs * gv.y + bv.y,
                            d.z * rs * gv.z + bv.z, d.w * rs * gv.w + bv.w);
    *(float4*)(out + (size_t)row * ldo + lane * 4) = ov;
}

// ---------------------------------------------------------------------------
// Flash attention for the fusion layer.  qkv: [SM, 384] rows = s*2+m.
// grid = (S/16 query tiles, M*H).  One wave; online softmax; dh = 32.
// Q and K fragments come straight from global; V and P go through LDS.
// ---------------------------------------------------------------------------
__global__ __launch_bounds__(32)
void flash_kernel(const float* __restrict__ qkv, float* __restrict__ ctxbuf, int S)
{
    __shared__ _Float16 Vs[32 * 32];
    __shared__ _Float16 Ps[16 * 32];
    const int lane = threadIdx.x;
    const int rl   = lane & 15;
    const int q0   = blockIdx.x * 16;
    const int m    = blockIdx.y >> 2;
    const int h    = blockIdx.y & 3;
    const float scale = 0.1767766952966369f;   // 1/sqrt(32)

    const float* qbase = qkv + (size_t)m * 384 + h * 32;        // Q at +0
    const float* kbase = qbase + 128;                           // K at +128
    const float* vbase = qbase + 256;                           // V at +256

    // Q fragment, pre-scaled (row stride between tokens = 2*384)
    v16h aq = load_a_frag(qbase + (size_t)q0 * 768, 768, lane, scale);

    FragF ctx0, ctx1;
    float mrow[8], lrow[8];
    #pragma unroll
    for (int v = 0; v < 8; ++v) { ctx0.f[v] = 0.f; ctx1.f[v] = 0.f; mrow[v] = -1e30f; lrow[v] = 0.f; }

    for (int t0 = 0; t0 < S; t0 += 32) {
        // stage 32 V rows into LDS as f16 (lane = key row, packed 16B stores)
        {
            const float* vrow = vbase + (size_t)(t0 + lane) * 768;
            #pragma unroll
            for (int c = 0; c < 4; ++c) {
                float4 w0 = *(const float4*)(vrow + c * 8);
                float4 w1 = *(const float4*)(vrow + c * 8 + 4);
                Pack8 pk;
                pk.h[0] = (_Float16)w0.x; pk.h[1] = (_Float16)w0.y;
                pk.h[2] = (_Float16)w0.z; pk.h[3] = (_Float16)w0.w;
                pk.h[4] = (_Float16)w1.x; pk.h[5] = (_Float16)w1.y;
                pk.h[6] = (_Float16)w1.z; pk.h[7] = (_Float16)w1.w;
                *(float4*)(&Vs[lane * 32 + c * 8]) = pk.q;
            }
        }
        __syncthreads();

        // scores for two 16-key tiles (QK^T): B fragment straight from global
        FragF sa, sb;
        #pragma unroll
        for (int v = 0; v < 8; ++v) { sa.f[v] = 0.f; sb.f[v] = 0.f; }
        {
            v16h bk0 = load_b_frag(kbase + (size_t)(t0)      * 768, 768, lane);
            v16h bk1 = load_b_frag(kbase + (size_t)(t0 + 16) * 768, 768, lane);
            sa.v = wmma_f16(aq, bk0, sa.v);
            sb.v = wmma_f16(aq, bk1, sb.v);
        }

        // online softmax (per-lane stats for 8 rows; reduce over 16 col-lanes)
        float pa[8], pb[8];
        #pragma unroll
        for (int v = 0; v < 8; ++v) {
            float mx = fmaxf(sa.f[v], sb.f[v]);
            #pragma unroll
            for (int off = 1; off < 16; off <<= 1) mx = fmaxf(mx, __shfl_xor(mx, off, 32));
            float mn = fmaxf(mrow[v], mx);
            float alpha = __expf(mrow[v] - mn);
            mrow[v] = mn;
            pa[v] = __expf(sa.f[v] - mn);
            pb[v] = __expf(sb.f[v] - mn);
            float ls = pa[v] + pb[v];
            #pragma unroll
            for (int off = 1; off < 16; off <<= 1) ls += __shfl_xor(ls, off, 32);
            lrow[v] = lrow[v] * alpha + ls;
            ctx0.f[v] *= alpha;
            ctx1.f[v] *= alpha;
        }

        // write P (C-layout -> row/col form)
        #pragma unroll
        for (int v = 0; v < 8; ++v) {
            int rr = v + ((lane >> 4) << 3);
            Ps[rr * 32 + rl]      = (_Float16)pa[v];
            Ps[rr * 32 + rl + 16] = (_Float16)pb[v];
        }
        __syncthreads();

        // A fragment of P: two contiguous 16B runs in LDS
        FragH ap;
        ap.q[0] = *(const float4*)(Ps + rl * 32 + ((lane >> 4) << 3));
        ap.q[1] = *(const float4*)(Ps + rl * 32 + 16 + ((lane >> 4) << 3));

        // B fragments of V (K-major -> strided LDS reads)
        FragH bv0, bv1;
        #pragma unroll
        for (int j = 0; j < 16; ++j) {
            int kk = (j & 7) + ((j >> 3) << 4) + ((lane >> 4) << 3);
            bv0.h[j] = Vs[kk * 32 + rl];
            bv1.h[j] = Vs[kk * 32 + 16 + rl];
        }
        ctx0.v = wmma_f16(ap.v, bv0.v, ctx0.v);
        ctx1.v = wmma_f16(ap.v, bv1.v, ctx1.v);
        __syncthreads();
    }

    #pragma unroll
    for (int v = 0; v < 8; ++v) {
        int row = q0 + v + ((lane >> 4) << 3);
        float il = 1.0f / lrow[v];
        size_t base = ((size_t)row * 2 + m) * DD + h * 32;
        ctxbuf[base + rl]      = ctx0.f[v] * il;
        ctxbuf[base + 16 + rl] = ctx1.f[v] * il;
    }
}

// mean over the 2 metapaths (float4 chunks)
__global__ void mean_kernel(const float* __restrict__ x2, float* __restrict__ out, int S)
{
    int t = blockIdx.x * blockDim.x + threadIdx.x;
    if (t >= S * 32) return;
    int s = t >> 5, c = (t & 31) * 4;
    float4 a = *(const float4*)(x2 + ((size_t)s * 2) * DD + c);
    float4 b = *(const float4*)(x2 + ((size_t)s * 2 + 1) * DD + c);
    float4 o = make_float4(0.5f * (a.x + b.x), 0.5f * (a.y + b.y),
                           0.5f * (a.z + b.z), 0.5f * (a.w + b.w));
    *(float4*)(out + (size_t)s * DD + c) = o;
}

// ---------------------------------------------------------------------------
// Host orchestration
// ---------------------------------------------------------------------------
extern "C" void kernel_launch(void* const* d_in, const int* in_sizes, int n_in,
                              void* d_out, int out_size, void* d_ws, size_t ws_size,
                              hipStream_t stream)
{
    const float* x_author = (const float*)d_in[0];
    const float* x_paper  = (const float*)d_in[1];
    const int*   src_ap   = (const int*)d_in[2];
    const int*   dst_ap   = (const int*)d_in[3];
    const int*   src_pa   = (const int*)d_in[4];
    const int*   dst_pa   = (const int*)d_in[5];
    const float* pa_W = (const float*)d_in[6];  const float* pa_b = (const float*)d_in[7];
    const float* pp_W = (const float*)d_in[8];  const float* pp_b = (const float*)d_in[9];
    const float* f_Wqkv = (const float*)d_in[10]; const float* f_bqkv = (const float*)d_in[11];
    const float* f_Wo   = (const float*)d_in[12]; const float* f_bo   = (const float*)d_in[13];
    const float* f_W1   = (const float*)d_in[14]; const float* f_b1   = (const float*)d_in[15];
    const float* f_W2   = (const float*)d_in[16]; const float* f_b2   = (const float*)d_in[17];
    const float* f_g1   = (const float*)d_in[18]; const float* f_bt1  = (const float*)d_in[19];
    const float* f_g2   = (const float*)d_in[20]; const float* f_bt2  = (const float*)d_in[21];
    const float* cls_W  = (const float*)d_in[22]; const float* cls_b  = (const float*)d_in[23];
    const float* lW[3][2]; const float* lb[3][2]; const float* lg[3]; const float* lbb[3];
    for (int i = 0; i < 3; ++i) {
        lW[i][0] = (const float*)d_in[24 + 6 * i + 0];  // Wv
        lb[i][0] = (const float*)d_in[24 + 6 * i + 1];  // bv
        lW[i][1] = (const float*)d_in[24 + 6 * i + 2];  // Wo
        lb[i][1] = (const float*)d_in[24 + 6 * i + 3];  // bo
        lg[i]    = (const float*)d_in[24 + 6 * i + 4];  // g
        lbb[i]   = (const float*)d_in[24 + 6 * i + 5];  // b
    }

    // ---- workspace layout (floats) ----
    float* ws = (float*)d_ws;
    size_t o = 0;
    float* hp    = ws + o; o += (size_t)NP * DD;      // later becomes t (in-place LN)
    float* proj  = ws + o; o += (size_t)NP * DD;      // |
    float* aggp  = ws + o; o += (size_t)NP * DD;      // | proj..aggp reused as FFN h1
    float* ha    = ws + o; o += (size_t)NA * DD;
    float* agga  = ws + o; o += (size_t)NA * DD;
    float* cnta  = ws + o; o += NA;
    float* cntp  = ws + o; o += NP;
    float* x     = ws + o; o += (size_t)SM * DD;      // stacked [NA,2,D]
    float* qkvb  = ws + o; o += (size_t)SM * 384;
    float* ctxb  = ws + o; o += (size_t)SM * DD;
    float* x1    = ws + o; o += (size_t)SM * DD;
    float* tmp   = ws + o; o += (size_t)SM * DD;      // attn-out, then FFN out
    float* x2    = ws + o; o += (size_t)SM * DD;
    float* meanb = ws + o; o += (size_t)NA * DD;
    float* Wf    = ws + o; o += 3 * DD * DD;
    float* bf    = ws + o; o += 3 * DD;
    float* h1    = proj;                              // [SM, DFF] spans proj+aggp

    // ---- fold metapath projections ----
    for (int i = 0; i < 3; ++i)
        fuse_w_kernel<<<DD, DD, 0, stream>>>(lW[i][0], lb[i][0], lW[i][1], lb[i][1],
                                             Wf + i * DD * DD, bf + i * DD);

    // ---- node feature projections ----
    gemm16_kernel<<<dim3(NA / 16, 2), 32, 0, stream>>>(x_author, 128, pa_W, pa_b, ha, DD, NA, DD, 128, 0);
    gemm16_kernel<<<dim3(NP / 16, 2), 32, 0, stream>>>(x_paper, 256, pp_W, pp_b, hp, DD, NP, DD, 256, 0);

    // ---- edge counts (cnta reused for both pa scatters) ----
    hipMemsetAsync(cnta, 0, NA * sizeof(float), stream);
    hipMemsetAsync(cntp, 0, NP * sizeof(float), stream);
    count_kernel<<<(EPA + 255) / 256, 256, 0, stream>>>(dst_pa, cnta, EPA);
    count_kernel<<<(EAP + 255) / 256, 256, 0, stream>>>(dst_ap, cntp, EAP);

    // ---- metapath 0: paper -> author ----
    gemm16_kernel<<<dim3(NP / 16, 2), 32, 0, stream>>>(hp, DD, Wf + 0 * DD * DD, bf + 0 * DD, proj, DD, NP, DD, DD, 0);
    hipMemsetAsync(agga, 0, (size_t)NA * DD * sizeof(float), stream);
    scatter_add_kernel<<<((size_t)EPA * 32 + 255) / 256, 256, 0, stream>>>(proj, src_pa, dst_pa, agga, EPA);
    ln_kernel<<<NA / 8, 256, 0, stream>>>(agga, DD, ha, DD, cnta, lg[0], lbb[0], x + 0, 2 * DD, NA);

    // ---- metapath 1 stage A: author -> paper (t, in place over hp) ----
    gemm16_kernel<<<dim3(NA / 16, 2), 32, 0, stream>>>(ha, DD, Wf + 1 * DD * DD, bf + 1 * DD, proj, DD, NA, DD, DD, 0);
    hipMemsetAsync(aggp, 0, (size_t)NP * DD * sizeof(float), stream);
    scatter_add_kernel<<<((size_t)EAP * 32 + 255) / 256, 256, 0, stream>>>(proj, src_ap, dst_ap, aggp, EAP);
    ln_kernel<<<NP / 8, 256, 0, stream>>>(aggp, DD, hp, DD, cntp, lg[1], lbb[1], hp, DD, NP);

    // ---- metapath 1 stage B: paper(t) -> author ----
    gemm16_kernel<<<dim3(NP / 16, 2), 32, 0, stream>>>(hp, DD, Wf + 2 * DD * DD, bf + 2 * DD, proj, DD, NP, DD, DD, 0);
    hipMemsetAsync(agga, 0, (size_t)NA * DD * sizeof(float), stream);
    scatter_add_kernel<<<((size_t)EPA * 32 + 255) / 256, 256, 0, stream>>>(proj, src_pa, dst_pa, agga, EPA);
    ln_kernel<<<NA / 8, 256, 0, stream>>>(agga, DD, ha, DD, cnta, lg[2], lbb[2], x + DD, 2 * DD, NA);

    // ---- fusion transformer layer ----
    gemm16_kernel<<<dim3(SM / 16, 6), 32, 0, stream>>>(x, DD, f_Wqkv, f_bqkv, qkvb, 384, SM, 384, DD, 0);
    flash_kernel<<<dim3(NA / 16, 8), 32, 0, stream>>>(qkvb, ctxb, NA);
    gemm16_kernel<<<dim3(SM / 16, 2), 32, 0, stream>>>(ctxb, DD, f_Wo, f_bo, tmp, DD, SM, DD, DD, 0);
    ln_kernel<<<SM / 8, 256, 0, stream>>>(tmp, DD, x, DD, nullptr, f_g1, f_bt1, x1, DD, SM);
    gemm16_kernel<<<dim3(SM / 16, DFF / 64), 32, 0, stream>>>(x1, DD, f_W1, f_b1, h1, DFF, SM, DFF, DD, 1);
    gemm16_kernel<<<dim3(SM / 16, 2), 32, 0, stream>>>(h1, DFF, f_W2, f_b2, tmp, DD, SM, DD, DFF, 0);
    ln_kernel<<<SM / 8, 256, 0, stream>>>(tmp, DD, x1, DD, nullptr, f_g2, f_bt2, x2, DD, SM);

    // ---- mean over metapaths + classifier ----
    mean_kernel<<<(NA * 32 + 255) / 256, 256, 0, stream>>>(x2, meanb, NA);
    gemm16_kernel<<<dim3(NA / 16, 1), 32, 0, stream>>>(meanb, DD, cls_W, cls_b, (float*)d_out, OUTD, NA, OUTD, DD, 0);

    (void)in_sizes; (void)n_in; (void)out_size; (void)ws_size;
}
